// DGF_Block_86844238725373
// MI455X (gfx1250) — matrix operand order repaired and checked by reading, not compiled, not run on hardware
//
#include <hip/hip_runtime.h>
#include <hip/hip_bf16.h>

// Problem constants (match reference)
#define BB 4
#define CCH 64
#define HH 96
#define WWD 320
#define HWP (HH * WWD)
#define EPSV 1e-5f

typedef __attribute__((ext_vector_type(16))) __bf16 v16bf;
typedef __attribute__((ext_vector_type(8)))  float  v8f;
typedef __attribute__((ext_vector_type(4)))  unsigned int v4u;
typedef __attribute__((ext_vector_type(8)))  int v8i;
typedef __attribute__((ext_vector_type(4)))  int v4i;

#if defined(__has_builtin)
#  if __has_builtin(__builtin_amdgcn_tensor_load_to_lds) && __has_builtin(__builtin_amdgcn_s_wait_tensorcnt)
#    define HAVE_TDM 1
#  else
#    define HAVE_TDM 0
#  endif
#else
#  define HAVE_TDM 0
#endif

// Native conversions: gfx1250 bf16 cvt hardware
__device__ __forceinline__ __bf16 f2bf(float f)  { return (__bf16)f; }
__device__ __forceinline__ float  bf2f(__bf16 h) { return (float)h; }
__device__ __forceinline__ int clampi(int v, int lo, int hi) {
    return v < lo ? lo : (v > hi ? hi : v);
}

// ---------------------------------------------------------------------------
// Weight prep: OIHW f32 -> [Opad][tap*Cin + c] bf16 (tap-major K order),
// with optional BN (gamma,beta,mean,var) folded into weights + bias.
// ---------------------------------------------------------------------------
__global__ void prep_weights(const float* __restrict__ w, const float* __restrict__ bsrc,
                             const float* __restrict__ g, const float* __restrict__ be,
                             const float* __restrict__ mn, const float* __restrict__ vr,
                             __bf16* __restrict__ wq, float* __restrict__ bq,
                             int O, int Opad, int Cin) {
    int tid = blockIdx.x * blockDim.x + threadIdx.x;
    int K = Cin * 9;
    if (tid >= Opad * K) return;
    int o  = tid / K;
    int fk = tid - o * K;
    int tap = fk / Cin;
    int c   = fk - tap * Cin;
    if (o < O) {
        float scale = 1.f, shift = 0.f;
        if (g) {
            float s = g[o] * rsqrtf(vr[o] + EPSV);
            scale = s;
            shift = be[o] - mn[o] * s;
        }
        wq[tid] = f2bf(w[(o * Cin + c) * 9 + tap] * scale);
        if (fk == 0) bq[o] = bsrc[o] * scale + shift;
    } else {
        wq[tid] = f2bf(0.f);
        if (fk == 0) bq[o] = 0.f;
    }
}

__global__ void cast_to_bf16(const float* __restrict__ in, __bf16* __restrict__ out, int n) {
    int i = blockIdx.x * blockDim.x + threadIdx.x;
    if (i < n) out[i] = f2bf(in[i]);
}

// ---------------------------------------------------------------------------
// Implicit-GEMM 3x3 conv via v_wmma_f32_16x16x32_bf16 with LDS staging.
// One workgroup = one 16-pixel tile; MT waves = MT 16-outch M-tiles.
// ---------------------------------------------------------------------------
template <int CIN, int MT, bool RELU_IN, bool RELU_OUT, bool OUT_BF16>
__global__ __launch_bounds__(32 * MT) void conv3x3_wmma(
    const __bf16* __restrict__ xa, const __bf16* __restrict__ xb,
    const __bf16* __restrict__ wq, const float* __restrict__ bq,
    void* __restrict__ outp, int coutStore) {
    constexpr int KTOT  = CIN * 9;
    constexpr int STEPS = KTOT / 32;
    constexpr int CP    = CIN + 4;          // padded channel stride (elements)

    __shared__ __bf16 lds[3 * 18 * CP];

    const int tid  = threadIdx.x;
    const int lane = tid & 31;
    const int wv   = tid >> 5;              // wave id = M tile

    int t = blockIdx.x;                     // pixel tile: 20 tiles per row
    const int tx = t % 20;  t /= 20;
    const int y  = t % HH;  t /= HH;
    const int b  = t;
    const int x0 = tx * 16;

    // ---- stage 3x18xCIN window: i = (c*3 + row)*18 + col (col fastest) ----
    constexpr int TOT = 3 * 18 * CIN;
    for (int i = tid; i < TOT; i += 32 * MT) {
        const int col = i % 18;
        const int rc  = i / 18;
        const int row = rc % 3;
        const int c   = rc / 3;
        const int yy = y + row - 1;
        const int xw = x0 + col - 1;
        const float vmul = ((yy >= 0) & (yy < HH) & (xw >= 0) & (xw < WWD)) ? 1.f : 0.f;
        const int yyc = clampi(yy, 0, HH - 1);
        const int xwc = clampi(xw, 0, WWD - 1);
        const __bf16* xp;
        int cc;
        if constexpr (CIN == 128) { xp = (c >= 64) ? xb : xa; cc = c & 63; }
        else                      { xp = xa;                  cc = c; }
        float f = bf2f(xp[((b * 64 + cc) * HH + yyc) * WWD + xwc]) * vmul;
        if (RELU_IN) f = fmaxf(f, 0.f);
        lds[(row * 18 + col) * CP + c] = f2bf(f);
    }
    __syncthreads();

    // ---- compute ----
    const int n  = lane & 15;
    const int hi = lane >> 4;
    const int xx = x0 + n;
    const int obase = wv * 16;

    v8f acc = {};
#pragma unroll
    for (int s = 0; s < STEPS; ++s) {
        // A fragment: weights, row M = obase + lane%16
        const __bf16* wrow = wq + (obase + n) * KTOT + s * 32 + hi * 16;
        v16bf af;
#pragma unroll
        for (int e = 0; e < 16; ++e) af[e] = wrow[e];

        // B fragment from LDS: col N = lane%16, K half by lane>=16
        const int kb  = s * 32 + hi * 16;   // 16-aligned => tap constant in run
        const int tap = kb / CIN;
        const int c0  = kb - tap * CIN;
        const int row = tap / 3;            // dy + 1
        const int col = n + tap % 3;        // n + dx + 1
        const __bf16* bp = &lds[(row * 18 + col) * CP + c0];
        v16bf bf;
#pragma unroll
        for (int j = 0; j < 16; ++j) bf[j] = bp[j];

        acc = __builtin_amdgcn_wmma_f32_16x16x32_bf16(false, af, false, bf,
                                                      (short)0, acc, false, false);
    }

    // ---- epilogue: D element r -> M = r + 8*hi, N = lane%16 ----
#pragma unroll
    for (int r = 0; r < 8; ++r) {
        const int o = obase + r + 8 * hi;
        float v = acc[r] + bq[o];
        if (RELU_OUT) v = fmaxf(v, 0.f);
        if (o < coutStore) {
            const int idx = ((b * coutStore + o) * HH + y) * WWD + xx;
            if constexpr (OUT_BF16) ((__bf16*)outp)[idx] = f2bf(v);
            else                    ((float*)outp)[idx]  = v;
        }
    }
}

// ---------------------------------------------------------------------------
// Modulated deformable conv, cooperative version: one workgroup (4 waves) per
// 16-pixel tile.
//   1) TDM (tensor_load_to_lds) pulls the 27x16 f32 head tile into LDS
//      (strided 2D tile, row stride H*W) -- s_wait_tensorcnt + barrier.
//   2) 144 workers derive per-(tap,pixel) bilinear corner offsets + weights.
//   3) All 128 threads build the 576x16 bf16 sampled-value tile in LDS
//      (each val = 4 masked corner gathers from feat_depth).
//   4) Each wave runs 18 WMMA steps: A = w_def fragment (global, L2-resident),
//      B = contiguous 16-element LDS rows (val padded to 584/row: conflict-free).
// Epilogue writes fd = depth + deform + bias as bf16 for the fusion conv.
// ---------------------------------------------------------------------------
#define VPAD 584   // 576 + 8: 8B-aligned rows, distinct banks across lanes

__global__ __launch_bounds__(128) void mdcn_wmma(
    const float* __restrict__ depth, const float* __restrict__ head,
    const __bf16* __restrict__ wq, const float* __restrict__ bq,
    __bf16* __restrict__ fd_bf) {
    __shared__ float  hlds[27 * 16];
    __shared__ int    a0l[144], a1l[144], a2l[144], a3l[144];
    __shared__ float  w0l[144], w1l[144], w2l[144], w3l[144];
    __shared__ __bf16 vall[16 * VPAD];

    const int tid  = threadIdx.x;
    const int lane = tid & 31;
    const int wv   = tid >> 5;

    int t = blockIdx.x;
    const int tx = t % 20;  t /= 20;
    const int y  = t % HH;  t /= HH;
    const int b  = t;
    const int x0 = tx * 16;

    // ---- 1) head tile -> LDS ----
    const float* hsrc = head + (size_t)(b * 27) * HWP + y * WWD + x0;
    if (wv == 0) {
#if HAVE_TDM
        // D# group0: count=1 | lds_addr | global_addr[56:0] | type=2
        const unsigned long long ga = (unsigned long long)(uintptr_t)hsrc;
        const unsigned ldsAddr = (unsigned)(uintptr_t)(&hlds[0]);
        v4u g0 = { 1u, ldsAddr, (unsigned)ga,
                   (unsigned)(((ga >> 32) & 0x1FFFFFFu) | (2u << 30)) };
        // D# group1: data_size=4B; tensor 16x27 (dim0 x dim1), tile 16x27,
        //            dim0_stride = H*W (channel-plane stride in elements)
        v8i g1 = { (int)(2u << 16),          // workgroup_mask=0, data_size=2 (4B)
                   (int)(16u << 16),         // tensor_dim0[15:0] in bits 63:48
                   (int)(27u << 16),         // tensor_dim0 hi=0, tensor_dim1=27
                   (int)(16u << 16),         // tensor_dim1 hi=0, tile_dim0=16
                   27,                       // tile_dim1=27, tile_dim2=0
                   HWP,                      // tensor_dim0_stride lo
                   0, 0 };                   // stride hi, tensor_dim1_stride=0
        v4i z4 = { 0, 0, 0, 0 };
#if defined(__clang_major__) && __clang_major__ >= 23
        v8i z8 = { 0, 0, 0, 0, 0, 0, 0, 0 };
        __builtin_amdgcn_tensor_load_to_lds(g0, g1, z4, z4, z8, 0);
#else
        __builtin_amdgcn_tensor_load_to_lds(g0, g1, z4, z4, 0);
#endif
        __builtin_amdgcn_s_wait_tensorcnt(0);
#else
        for (int i = lane; i < 27 * 16; i += 32)
            hlds[i] = hsrc[(size_t)(i / 16) * HWP + (i % 16)];
#endif
    }
    __syncthreads();

    // ---- 2) per-(tap,pixel) bilinear corner offsets + mask-folded weights ----
    for (int i = tid; i < 144; i += 128) {
        const int kk = i / 16, np = i % 16;
        const int ky = kk / 3 - 1, kx = kk % 3 - 1;
        const float dyv = hlds[(2 * kk) * 16 + np];
        const float dxv = hlds[(2 * kk + 1) * 16 + np];
        const float m   = 1.f / (1.f + __expf(-hlds[(18 + kk) * 16 + np]));
        const float py = (float)(y + ky) + dyv;
        const float px = (float)(x0 + np + kx) + dxv;
        const float fy = floorf(py), fx = floorf(px);
        const float wy = py - fy,    wx = px - fx;
        const int y0 = (int)fy, x0i = (int)fx;
        const int y1 = y0 + 1,  x1i = x0i + 1;
        const float oy = 1.f - wy, ox = 1.f - wx;
        const bool iy0 = (y0 >= 0) & (y0 < HH), iy1 = (y1 >= 0) & (y1 < HH);
        const bool ix0 = (x0i >= 0) & (x0i < WWD), ix1 = (x1i >= 0) & (x1i < WWD);
        const int y0c = clampi(y0, 0, HH - 1), y1c = clampi(y1, 0, HH - 1);
        const int x0c = clampi(x0i, 0, WWD - 1), x1c = clampi(x1i, 0, WWD - 1);
        a0l[i] = y0c * WWD + x0c;  w0l[i] = (iy0 & ix0) ? oy * ox * m : 0.f;
        a1l[i] = y0c * WWD + x1c;  w1l[i] = (iy0 & ix1) ? oy * wx * m : 0.f;
        a2l[i] = y1c * WWD + x0c;  w2l[i] = (iy1 & ix0) ? wy * ox * m : 0.f;
        a3l[i] = y1c * WWD + x1c;  w3l[i] = (iy1 & ix1) ? wy * wx * m : 0.f;
    }
    __syncthreads();

    // ---- 3) sampled-value tile: val[n][flatk], flatk = tap*64 + c ----
    for (int i = tid; i < 576 * 16; i += 128) {
        const int flatk = i >> 4, nv = i & 15;
        const int tap = flatk >> 6, c = flatk & 63;
        const int pi = tap * 16 + nv;
        const float* p = depth + (size_t)(b * 64 + c) * HWP;
        const float v = w0l[pi] * p[a0l[pi]] + w1l[pi] * p[a1l[pi]] +
                        w2l[pi] * p[a2l[pi]] + w3l[pi] * p[a3l[pi]];
        vall[nv * VPAD + flatk] = f2bf(v);
    }
    __syncthreads();

    // ---- 4) WMMA: 4 waves x (16 outch x 16 px), K = 576 ----
    const int n  = lane & 15;
    const int hi = lane >> 4;
    const int xx = x0 + n;
    const int obase = wv * 16;

    v8f acc = {};
#pragma unroll
    for (int s = 0; s < 18; ++s) {
        const __bf16* wrow = wq + (obase + n) * 576 + s * 32 + hi * 16;
        v16bf af;
#pragma unroll
        for (int e = 0; e < 16; ++e) af[e] = wrow[e];

        const __bf16* bp = &vall[n * VPAD + s * 32 + hi * 16];
        v16bf bf;
#pragma unroll
        for (int j = 0; j < 16; ++j) bf[j] = bp[j];

        acc = __builtin_amdgcn_wmma_f32_16x16x32_bf16(false, af, false, bf,
                                                      (short)0, acc, false, false);
    }

#pragma unroll
    for (int r = 0; r < 8; ++r) {
        const int o = obase + r + 8 * hi;
        const int idx = ((b * 64 + o) * HH + y) * WWD + xx;
        fd_bf[idx] = f2bf(depth[idx] + acc[r] + bq[o]);
    }
}

// ---------------------------------------------------------------------------
extern "C" void kernel_launch(void* const* d_in, const int* in_sizes, int n_in,
                              void* d_out, int out_size, void* d_ws, size_t ws_size,
                              hipStream_t stream) {
    const float* feat_rgb   = (const float*)d_in[0];
    const float* feat_depth = (const float*)d_in[1];
    const float* w_off1 = (const float*)d_in[2];
    const float* b_off1 = (const float*)d_in[3];
    const float* g1  = (const float*)d_in[4];
    const float* be1 = (const float*)d_in[5];
    const float* m1  = (const float*)d_in[6];
    const float* v1  = (const float*)d_in[7];
    const float* w_off2 = (const float*)d_in[8];
    const float* b_off2 = (const float*)d_in[9];
    const float* w_def  = (const float*)d_in[10];
    const float* b_def  = (const float*)d_in[11];
    const float* w_f1   = (const float*)d_in[12];
    const float* b_f1   = (const float*)d_in[13];
    const float* g2  = (const float*)d_in[14];
    const float* be2 = (const float*)d_in[15];
    const float* m2  = (const float*)d_in[16];
    const float* v2  = (const float*)d_in[17];
    const float* w_out = (const float*)d_in[18];
    const float* b_out = (const float*)d_in[19];

    const int NE = BB * CCH * HH * WWD;       // 7,864,320
    const int NH = BB * 27 * HH * WWD;        // head elements

    // workspace carve-out (256B aligned)
    char* base = (char*)d_ws;
    size_t off = 0;
    auto take = [&](size_t bytes) -> char* {
        char* p = base + off;
        off += (bytes + 255) & ~(size_t)255;
        return p;
    };
    __bf16* rgb_bf = (__bf16*)take((size_t)NE * 2);
    __bf16* t_bf   = (__bf16*)take((size_t)NE * 2);
    __bf16* fd_bf  = (__bf16*)take((size_t)NE * 2);
    __bf16* h_bf   = (__bf16*)take((size_t)NE * 2);
    float*  head   = (float*)take((size_t)NH * 4);
    __bf16* wq1 = (__bf16*)take(64 * 576 * 2);   float* bq1 = (float*)take(64 * 4);
    __bf16* wq2 = (__bf16*)take(32 * 576 * 2);   float* bq2 = (float*)take(32 * 4);
    __bf16* wqd = (__bf16*)take(64 * 576 * 2);   float* bqd = (float*)take(64 * 4);
    __bf16* wqf = (__bf16*)take(64 * 1152 * 2);  float* bqf = (float*)take(64 * 4);
    __bf16* wqo = (__bf16*)take(16 * 576 * 2);   float* bqo = (float*)take(16 * 4);

    auto cdiv = [](int a, int b) { return (a + b - 1) / b; };

    // weight prep (BN folded where present)
    prep_weights<<<cdiv(64 * 576, 256), 256, 0, stream>>>(w_off1, b_off1, g1, be1, m1, v1, wq1, bq1, 64, 64, 64);
    prep_weights<<<cdiv(32 * 576, 256), 256, 0, stream>>>(w_off2, b_off2, nullptr, nullptr, nullptr, nullptr, wq2, bq2, 27, 32, 64);
    prep_weights<<<cdiv(64 * 576, 256), 256, 0, stream>>>(w_def, b_def, nullptr, nullptr, nullptr, nullptr, wqd, bqd, 64, 64, 64);
    prep_weights<<<cdiv(64 * 1152, 256), 256, 0, stream>>>(w_f1, b_f1, g2, be2, m2, v2, wqf, bqf, 64, 64, 128);
    prep_weights<<<cdiv(16 * 576, 256), 256, 0, stream>>>(w_out, b_out, nullptr, nullptr, nullptr, nullptr, wqo, bqo, 1, 16, 64);
    cast_to_bf16<<<cdiv(NE, 256), 256, 0, stream>>>(feat_rgb, rgb_bf, NE);

    const int NTILES = BB * HH * (WWD / 16);  // 7680 pixel tiles

    // conv1 + BN1 -> t (bf16): 4 M-tile waves per block
    conv3x3_wmma<64, 4, false, false, true><<<NTILES, 128, 0, stream>>>(
        rgb_bf, nullptr, wq1, bq1, (void*)t_bf, 64);
    // conv2 -> head (f32, 27 ch): 2 M-tile waves per block
    conv3x3_wmma<64, 2, false, false, false><<<NTILES, 64, 0, stream>>>(
        t_bf, nullptr, wq2, bq2, (void*)head, 27);
    // modulated deformable conv + residual -> fd (bf16): cooperative 4-wave blocks
    mdcn_wmma<<<NTILES, 128, 0, stream>>>(feat_depth, head, wqd, bqd, fd_bf);
    // relu(concat(fd, rgb)) -> conv f1 + BN2 + relu -> h (bf16)
    conv3x3_wmma<128, 4, true, true, true><<<NTILES, 128, 0, stream>>>(
        fd_bf, rgb_bf, wqf, bqf, (void*)h_bf, 64);
    // final conv -> d_out (f32, 1 ch)
    conv3x3_wmma<64, 1, false, false, false><<<NTILES, 32, 0, stream>>>(
        h_bf, nullptr, wqo, bqo, d_out, 1);

    (void)in_sizes; (void)n_in; (void)out_size; (void)ws_size;
}
